// WinMSA_87428354277382
// MI455X (gfx1250) — matrix operand order repaired and checked by reading, not compiled, not run on hardware
//
#include <hip/hip_runtime.h>

typedef __attribute__((ext_vector_type(8)))  _Float16 v8h;
typedef __attribute__((ext_vector_type(16))) _Float16 v16h;
typedef __attribute__((ext_vector_type(8)))  float    v8f;
typedef __attribute__((ext_vector_type(4)))  unsigned int u32x4;
typedef __attribute__((ext_vector_type(4)))  int      i32x4;
typedef __attribute__((ext_vector_type(8)))  int      i32x8;

#define S_VALID 49
#define C_EMB   192
#define HEADS   6
#define NWIN    4096

// LDS strides in halfs (all row strides are multiples of 8 halfs = 16B)
#define XS_STRIDE 192   // X / O: 64 x 192
#define QS_STRIDE 40    // Q: per head 64 x 32 (+8 pad) -> conflict-free A reads
#define KS_STRIDE 32    // K: per head 64 x 32
#define VT_STRIDE 72    // V^T: per head 32 x 64 (+8 pad)
#define PS_STRIDE 72    // P: per head 64 x 64 (+8 pad)

#define XS_OFF 0
#define QS_OFF (XS_OFF + 64 * XS_STRIDE)
#define KS_OFF (QS_OFF + HEADS * 64 * QS_STRIDE)
#define VT_OFF (KS_OFF + HEADS * 64 * KS_STRIDE)
#define PS_OFF (VT_OFF + HEADS * 32 * VT_STRIDE)
#define SMEM_HALFS (PS_OFF + HEADS * 64 * PS_STRIDE)
#define SMEM_BYTES (SMEM_HALFS * 2)   // 162816 B < 320 KB, ~2 WG/WGP

// ---------------- fragment helpers (contiguous k order, lane half h holds k 16h..16h+15) --

// A fragment (16x32 f16, M x K) from row-major LDS [m][k].
// lanes 0-15: row m=l, halfs k = kbase+0..15
// lanes 16-31: row m=l-16, halfs k = kbase+16..31
__device__ __forceinline__ v16h load_frag_A(const _Float16* m, int row0,
                                            int stride, int kbase, int lane) {
  int r   = row0 + (lane & 15);
  int off = kbase + ((lane >> 4) << 4);
  const _Float16* p = m + r * stride + off;
  v8h lo = *(const v8h*)(p);
  v8h hi = *(const v8h*)(p + 8);
  return __builtin_shufflevector(lo, hi, 0, 1, 2, 3, 4, 5, 6, 7,
                                 8, 9, 10, 11, 12, 13, 14, 15);
}

// B fragment (32x16 f16, K x N) from n-major LDS storage [n][k].
// lanes 0-15: col n=l, k = kbase+0..15 ; lanes 16-31: col n=l-16, k = kbase+16..31
__device__ __forceinline__ v16h load_frag_B(const _Float16* m, int col0,
                                            int stride, int kbase, int lane) {
  int n   = col0 + (lane & 15);
  int off = kbase + ((lane >> 4) << 4);
  const _Float16* p = m + n * stride + off;
  v8h lo = *(const v8h*)(p);
  v8h hi = *(const v8h*)(p + 8);
  return __builtin_shufflevector(lo, hi, 0, 1, 2, 3, 4, 5, 6, 7,
                                 8, 9, 10, 11, 12, 13, 14, 15);
}

__device__ __forceinline__ v8f wmma_f16(v16h a, v16h b, v8f c) {
  return __builtin_amdgcn_wmma_f32_16x16x32_f16(false, a, false, b,
                                                (short)0, c, false, false);
}

// ------------------------- prepack kernels -----------------------------------

// Pack fp32 row-major weight W (K x N) into per-tile B-fragment order:
// tile t = kt*(N/16)+nt; per lane l, 16 contiguous halfs e=0..15 holding
// W[kt*32 + (l<16?0:16) + e][nt*16 + (l&15)].
__global__ void winmsa_pack_w(const float* __restrict__ W,
                              _Float16* __restrict__ out, int K, int N) {
  int t = blockIdx.x, l = threadIdx.x;
  int NT = N >> 4;
  int kt = t / NT, nt = t % NT;
  int col = nt * 16 + (l & 15);
  int kb  = kt * 32 + ((l >> 4) << 4);
  _Float16* o = out + ((size_t)t * 32 + l) * 16;
#pragma unroll
  for (int e = 0; e < 16; ++e) o[e] = (_Float16)W[(size_t)(kb + e) * N + col];
}

// Gather Swin relative-position bias into biasG[h][q][k], q,k in [0,49)
__global__ void winmsa_pack_bias(const float* __restrict__ table,
                                 float* __restrict__ biasG) {
  int i = blockIdx.x * blockDim.x + threadIdx.x;
  if (i >= HEADS * S_VALID * S_VALID) return;
  int h = i / (S_VALID * S_VALID);
  int r = i % (S_VALID * S_VALID);
  int q = r / S_VALID, k = r % S_VALID;
  int dy = (k / 7) - (q / 7) + 6;
  int dx = (k % 7) - (q % 7) + 6;
  biasG[i] = table[(dy * 13 + dx) * HEADS + h];
}

// ------------------------- main attention kernel -----------------------------

__global__ __launch_bounds__(256)
void winmsa_main(const float* __restrict__ X, const _Float16* __restrict__ wqkvP,
                 const float* __restrict__ bqkv, const _Float16* __restrict__ woP,
                 const float* __restrict__ bo, const float* __restrict__ biasG,
                 float* __restrict__ out) {
  extern __shared__ __align__(16) char smem_raw[];
  _Float16* smem = (_Float16*)smem_raw;
  _Float16* Xs = smem + XS_OFF;   // also reused as O after stage 3
  _Float16* Qs = smem + QS_OFF;
  _Float16* Ks = smem + KS_OFF;
  _Float16* Vt = smem + VT_OFF;
  _Float16* Ps = smem + PS_OFF;

  const int tid  = threadIdx.x;
  const int lane = tid & 31;
  const int wave = tid >> 5;

  // ---- stage 0: X window -> LDS f16, pad rows [49,64) with zeros ------------
  // Preferred path: Tensor Data Mover async-copies the raw fp32 tile into an
  // LDS staging area (aliases Ps, first written 2 barriers later), then VALU
  // converts to f16. Fallback: plain global loads + convert.
  const float* Xw = X + (size_t)blockIdx.x * S_VALID * C_EMB;
#if __has_builtin(__builtin_amdgcn_tensor_load_to_lds)
  if (wave == 0) {
    unsigned long long ga = (unsigned long long)(size_t)Xw;
    u32x4 g0;
    g0.x = 1u;                                    // count=1, user descriptor
    g0.y = (unsigned)(PS_OFF * 2);                // lds_addr (bytes)
    g0.z = (unsigned)(ga & 0xFFFFFFFFu);          // global_addr[31:0]
    g0.w = (unsigned)((ga >> 32) & 0x01FFFFFFu)   // global_addr[56:32]
           | (2u << 30);                          // type=2 ("image")
    i32x8 g1;
    g1[0] = (int)(2u << 16);                      // wg_mask=0, data_size=4B
    g1[1] = (int)(192u << 16);                    // tensor_dim0 lo16 (=192)
    g1[2] = (int)(49u << 16);                     // dim0 hi16=0, tensor_dim1=49
    g1[3] = (int)(192u << 16);                    // dim1 hi16=0, tile_dim0=192
    g1[4] = (int)(49u);                           // tile_dim1=49, tile_dim2=0
    g1[5] = (int)(192u);                          // tensor_dim0_stride lo32
    g1[6] = 0;                                    // stride hi16, dim1_stride lo
    g1[7] = 0;
    i32x4 z4 = {0, 0, 0, 0};                      // groups 2/3 unused (2-D)
#if __has_include(<hip/amd_detail/amd_gfx1250_TDM.h>)
    i32x8 z8 = {0, 0, 0, 0, 0, 0, 0, 0};
    __builtin_amdgcn_tensor_load_to_lds(g0, g1, z4, z4, z8, 0);
#else
    __builtin_amdgcn_tensor_load_to_lds(g0, g1, z4, z4, 0);
#endif
    __builtin_amdgcn_s_wait_tensorcnt((short)0);
  }
  __syncthreads();
  {
    const float* stg = (const float*)(smem_raw + PS_OFF * 2);
    for (int i = tid; i < 64 * C_EMB; i += 256)
      Xs[i] = (i < S_VALID * C_EMB) ? (_Float16)stg[i] : (_Float16)0.0f;
  }
#else
  for (int i = tid; i < 64 * C_EMB; i += 256)
    Xs[i] = (i < S_VALID * C_EMB) ? (_Float16)Xw[i] : (_Float16)0.0f;
#endif
  __syncthreads();

  // ---- stage 1: QKV = X @ Wqkv + bqkv  (64x576, K=192) ----------------------
  // 4 M-tiles x 36 N-tiles = 144 tiles, 18 per wave
  for (int t = wave; t < 144; t += 8) {
    int mt = t / 36, nt = t % 36;
    v8f acc = {0.f, 0.f, 0.f, 0.f, 0.f, 0.f, 0.f, 0.f};
#pragma unroll
    for (int kt = 0; kt < 6; ++kt) {
      v16h a = load_frag_A(Xs, mt * 16, XS_STRIDE, kt * 32, lane);
      v16h b = *(const v16h*)(wqkvP + ((size_t)(kt * 36 + nt) * 32 + lane) * 16);
      acc = wmma_f16(a, b, acc);
    }
    int c3 = nt * 16 + (lane & 15);          // 0..575
    float bq = bqkv[c3];
    int m0 = mt * 16 + ((lane >> 4) << 3);   // D rows: 0..7 (lo lanes) / 8..15
    int ntu = __builtin_amdgcn_readfirstlane(nt);  // wave-uniform -> scalar branch
    if (ntu < 12) {                          // Q -> row-major [h][s][d]
      int h = c3 >> 5, d = c3 & 31;
      _Float16* q = Qs + (size_t)h * 64 * QS_STRIDE + d;
#pragma unroll
      for (int r = 0; r < 8; ++r) q[(m0 + r) * QS_STRIDE] = (_Float16)(acc[r] + bq);
    } else if (ntu < 24) {                   // K -> row-major [h][s][d]
      int ck = c3 - 192, h = ck >> 5, d = ck & 31;
      _Float16* kk = Ks + (size_t)h * 64 * KS_STRIDE + d;
#pragma unroll
      for (int r = 0; r < 8; ++r) kk[(m0 + r) * KS_STRIDE] = (_Float16)(acc[r] + bq);
    } else {                                 // V -> transposed [h][d][s] (contig)
      int cv = c3 - 384, h = cv >> 5, d = cv & 31;
      _Float16* vp = Vt + ((size_t)h * 32 + d) * VT_STRIDE + m0;
      v8h pk;
#pragma unroll
      for (int r = 0; r < 8; ++r) pk[r] = (_Float16)(acc[r] + bq);
      *(v8h*)vp = pk;
    }
  }
  __syncthreads();

  // ---- stage 2: attn = scale*Q K^T + bias, fused in-register softmax --------
  // 6 heads x 4 M-tiles = 24 row-blocks, 3 per wave; each block = 4 WMMAs
  const float scale = 0.17677669529663687f;  // 32^-0.5
  for (int rb = wave; rb < 24; rb += 8) {
    int h = rb >> 2, mt = rb & 3;
    const _Float16* Qh = Qs + (size_t)h * 64 * QS_STRIDE;
    const _Float16* Kh = Ks + (size_t)h * 64 * KS_STRIDE;
    v16h a = load_frag_A(Qh, mt * 16, QS_STRIDE, 0, lane);
    v8f acc[4];
#pragma unroll
    for (int nt = 0; nt < 4; ++nt) {
      v16h b = load_frag_B(Kh, nt * 16, KS_STRIDE, 0, lane);
      v8f z = {0.f, 0.f, 0.f, 0.f, 0.f, 0.f, 0.f, 0.f};
      acc[nt] = wmma_f16(a, b, z);
    }
    int colbase = lane & 15;
    int rowoff  = (lane >> 4) << 3;
#pragma unroll
    for (int r = 0; r < 8; ++r) {
      int q = mt * 16 + rowoff + r;
      float v[4], mx = -1.0e30f;
#pragma unroll
      for (int t = 0; t < 4; ++t) {
        int col = t * 16 + colbase;
        float bb = (q < S_VALID && col < S_VALID)
                       ? biasG[((size_t)h * S_VALID + q) * S_VALID + col] : 0.f;
        float val = (col < S_VALID) ? (acc[t][r] * scale + bb) : -1.0e30f;
        v[t] = val;
        mx = fmaxf(mx, val);
      }
      // row reductions across the 16-lane half-groups (wave32)
#pragma unroll
      for (int msk = 1; msk < 16; msk <<= 1) mx = fmaxf(mx, __shfl_xor(mx, msk, 32));
      float s = 0.f;
#pragma unroll
      for (int t = 0; t < 4; ++t) { v[t] = __expf(v[t] - mx); s += v[t]; }
#pragma unroll
      for (int msk = 1; msk < 16; msk <<= 1) s += __shfl_xor(s, msk, 32);
      float inv = 1.f / s;
      _Float16* prow = Ps + ((size_t)h * 64 + q) * PS_STRIDE;
#pragma unroll
      for (int t = 0; t < 4; ++t) {
        int col = t * 16 + colbase;
        prow[col] = (_Float16)((col < S_VALID) ? v[t] * inv : 0.f);
      }
    }
  }
  __syncthreads();

  // ---- stage 3: O = P @ V  (per head 64x32, K=64) -> reuse Xs as O ----------
  // 6 heads x 4 M-tiles x 2 N-tiles = 48 tiles, 6 per wave
  for (int t = wave; t < 48; t += 8) {
    int h = t >> 3, rem = t & 7, mt = rem >> 1, nt = rem & 1;
    const _Float16* Ph = Ps + (size_t)h * 64 * PS_STRIDE;
    const _Float16* Vh = Vt + (size_t)h * 32 * VT_STRIDE;
    v8f acc = {0.f, 0.f, 0.f, 0.f, 0.f, 0.f, 0.f, 0.f};
#pragma unroll
    for (int kt = 0; kt < 2; ++kt) {
      v16h a = load_frag_A(Ph, mt * 16, PS_STRIDE, kt * 32, lane);
      v16h b = load_frag_B(Vh, nt * 16, VT_STRIDE, kt * 32, lane);
      acc = wmma_f16(a, b, acc);
    }
    int c  = h * 32 + nt * 16 + (lane & 15);
    int m0 = mt * 16 + ((lane >> 4) << 3);
    _Float16* o = Xs + c;   // O row-major [s][c], stride 192
#pragma unroll
    for (int r = 0; r < 8; ++r) o[(m0 + r) * XS_STRIDE] = (_Float16)acc[r];
  }
  __syncthreads();

  // ---- stage 4: Y = O @ Wo + bo  (49x192, K=192) ----------------------------
  // 4 M-tiles x 12 N-tiles = 48 tiles, 6 per wave
  float* outw = out + (size_t)blockIdx.x * S_VALID * C_EMB;
  for (int t = wave; t < 48; t += 8) {
    int mt = t / 12, nt = t % 12;
    v8f acc = {0.f, 0.f, 0.f, 0.f, 0.f, 0.f, 0.f, 0.f};
#pragma unroll
    for (int kt = 0; kt < 6; ++kt) {
      v16h a = load_frag_A(Xs, mt * 16, XS_STRIDE, kt * 32, lane);
      v16h b = *(const v16h*)(woP + ((size_t)(kt * 12 + nt) * 32 + lane) * 16);
      acc = wmma_f16(a, b, acc);
    }
    int c  = nt * 16 + (lane & 15);
    float bb = bo[c];
    int m0 = mt * 16 + ((lane >> 4) << 3);
#pragma unroll
    for (int r = 0; r < 8; ++r) {
      int s = m0 + r;
      if (s < S_VALID) outw[(size_t)s * C_EMB + c] = acc[r] + bb;
    }
  }
}

// ------------------------------ launcher -------------------------------------

extern "C" void kernel_launch(void* const* d_in, const int* in_sizes, int n_in,
                              void* d_out, int out_size, void* d_ws, size_t ws_size,
                              hipStream_t stream) {
  (void)in_sizes; (void)n_in; (void)out_size; (void)ws_size;
  const float* X      = (const float*)d_in[0];
  const float* Wqkv   = (const float*)d_in[1];
  const float* bqkv   = (const float*)d_in[2];
  const float* Wo     = (const float*)d_in[3];
  const float* bo     = (const float*)d_in[4];
  const float* btable = (const float*)d_in[5];
  float* out = (float*)d_out;

  // workspace layout: packed Wqkv tiles | packed Wo tiles | gathered bias
  _Float16* wqkvP = (_Float16*)d_ws;                              // 216 KB
  _Float16* woP   = (_Float16*)((char*)d_ws + 216 * 1024);        // 72 KB
  float*    biasG = (float*)((char*)d_ws + (216 + 72) * 1024);    // ~57 KB

  winmsa_pack_w<<<6 * 36, 32, 0, stream>>>(Wqkv, wqkvP, 192, 576);
  winmsa_pack_w<<<6 * 12, 32, 0, stream>>>(Wo, woP, 192, 192);
  winmsa_pack_bias<<<(HEADS * S_VALID * S_VALID + 255) / 256, 256, 0, stream>>>(
      btable, biasG);

  hipFuncSetAttribute(reinterpret_cast<const void*>(winmsa_main),
                      hipFuncAttributeMaxDynamicSharedMemorySize, SMEM_BYTES);
  winmsa_main<<<NWIN, 256, SMEM_BYTES, stream>>>(X, wqkvP, bqkv, woP, bo, biasG,
                                                 out);
}
